// MSAFMamba_37417755082959
// MI455X (gfx1250) — matrix operand order repaired and checked by reading, not compiled, not run on hardware
//
#include <hip/hip_runtime.h>
#include <math.h>

// ---------------------------------------------------------------------------
// MSAF-Mamba forward for MI455X (gfx1250), wave32 + WMMA f32_16x16x32_f16.
//
// Input flattening assumption (JAX pytree order: dict keys sorted, lists in
// order), n_in == 150:
//   0 af_b1, 1 af_b2, 2 af_w1, 3 af_w2,
//   blocks[k] at base bb = 4 + 27*k:
//     +0 fuse_w, +1 sc_b, +2 sc_ln_b, +3 sc_ln_g, +4 sc_w,
//     spa at bb+5, spe at bb+16, each ssm dict:
//       +0 A_log, +1 D, +2 conv_b, +3 conv_w, +4 dt_b, +5 dt_w,
//       +6 in_proj, +7 ln_b, +8 ln_g, +9 out_proj, +10 x_proj
//   fe[s] at base 85 + 14*s:
//     +0 bn1.b, +1 bn1.g, +2 bn2.b, +3 bn2.g, +4 bn3.b, +5 bn3.g,
//     +6 c3a_b, +7 c3a_w, +8 c3b_b, +9 c3b_w, +10 dw_b, +11 dw_w,
//     +12 pw_b, +13 pw_w
//   141 fu_b, 142 fu_bn.b, 143 fu_bn.g, 144 fu_w,
//   145 head_b, 146 head_w, 147 norm_b, 148 norm_g, 149 x
// Output: (64, 16) float32.
// ---------------------------------------------------------------------------

#define BB     64
#define CIN    30
#define HH     15
#define HW     225      // 15*15 == L
#define LL     225
#define EE     128
#define DI     256
#define DSTATE 16
#define DTRANK 8
#define NC     16
#define BL     (BB*LL)  // 14400

typedef __attribute__((ext_vector_type(16))) _Float16 v16h;
typedef __attribute__((ext_vector_type(8)))  _Float16 v8h;
typedef __attribute__((ext_vector_type(8)))  float    v8f;

static __device__ __forceinline__ float sigf(float x) {
  return 1.0f / (1.0f + __expf(-x));
}

// ---------------------------------------------------------------------------
// Weight pre-pack: fp32 (N,K) row-major -> f16 (Npad,Kpad) zero-padded.
// Makes every B-fragment load an unconditional 16B global_load_b128.
// ---------------------------------------------------------------------------
__global__ void pack_w_f16(const float* __restrict__ Wsrc, _Float16* __restrict__ dst,
                           int N, int K, int Kpad, int total) {
  const int idx = blockIdx.x * blockDim.x + threadIdx.x;
  if (idx >= total) return;
  const int n = idx / Kpad, k = idx % Kpad;
  dst[idx] = (n < N && k < K) ? (_Float16)Wsrc[(size_t)n * K + k] : (_Float16)0.0f;
}

// A fragment (fp32 -> f16): lanes 0-15 rows, halves = K groups
// {0..7,16..23}; lanes 16-31 -> {8..15,24..31}. Unconditional float4 loads.
static __device__ __forceinline__ v16h load_a(const float* __restrict__ Arow,
                                              int k0, int hi) {
  const float4 A0 = *(const float4*)(Arow + k0 + hi * 8);
  const float4 A1 = *(const float4*)(Arow + k0 + hi * 8 + 4);
  const float4 A2 = *(const float4*)(Arow + k0 + 16 + hi * 8);
  const float4 A3 = *(const float4*)(Arow + k0 + 16 + hi * 8 + 4);
  v16h av;
  av[0]  = (_Float16)A0.x; av[1]  = (_Float16)A0.y; av[2]  = (_Float16)A0.z; av[3]  = (_Float16)A0.w;
  av[4]  = (_Float16)A1.x; av[5]  = (_Float16)A1.y; av[6]  = (_Float16)A1.z; av[7]  = (_Float16)A1.w;
  av[8]  = (_Float16)A2.x; av[9]  = (_Float16)A2.y; av[10] = (_Float16)A2.z; av[11] = (_Float16)A2.w;
  av[12] = (_Float16)A3.x; av[13] = (_Float16)A3.y; av[14] = (_Float16)A3.z; av[15] = (_Float16)A3.w;
  return av;
}

// B fragment from padded f16 weights: lane = column, halves = K groups
// {0..7,16..23}; lanes 16-31 -> {8..15,24..31}. Two unconditional 16B loads.
static __device__ __forceinline__ v16h load_b(const _Float16* __restrict__ Brow,
                                              int k0, int hi) {
  const v8h b0 = *(const v8h*)(Brow + k0 + hi * 8);
  const v8h b1 = *(const v8h*)(Brow + k0 + 16 + hi * 8);
  v16h bv;
#pragma unroll
  for (int i = 0; i < 8; ++i) { bv[i] = b0[i]; bv[i + 8] = b1[i]; }
  return bv;
}

static __device__ __forceinline__ v8f wmma16(v16h a, v16h b, v8f c) {
  return __builtin_amdgcn_wmma_f32_16x16x32_f16(
      /*neg_a=*/false, a, /*neg_b=*/false, b,
      /*c_mod=*/(short)0, c, /*reuse_a=*/false, /*reuse_b=*/false);
}

// ---------------------------------------------------------------------------
// GEMM: C[M,N] = act( bn( A[M,K] * Wpad^T + bias ) ).  M % 16 == 0 always
// (14400 or 64 here), Kpad % 32 == 0. Each wave owns a 16x64 tile (4 accs
// share one A fragment -> 4x less A traffic/convert, 8 WMMAs per loop body);
// ragged Npad (48/16) takes a uniform fallback. Only stores are guarded.
// act: 0 none, 1 relu, 2 softplus, 3 sigmoid. Block (32,4) = 4 wave32.
// ---------------------------------------------------------------------------
__global__ void gemm_wmma_f16(const float* __restrict__ A, int lda,
                              const _Float16* __restrict__ Wh, int Kpad,
                              const float* __restrict__ bias,
                              const float* __restrict__ bng,
                              const float* __restrict__ bnb,
                              float bninv,
                              float* __restrict__ C, int ldc,
                              int M, int N, int Npad, int act) {
  const int lane = threadIdx.x, hi = lane >> 4, lm = lane & 15;
  const int m0 = (blockIdx.y * 4 + threadIdx.y) * 16;
  const int n0 = blockIdx.x * 64;
  if (m0 >= M) return;                       // wave-uniform
  const int rem = (Npad - n0) >> 4;
  const int ntiles = rem < 4 ? rem : 4;      // wave-uniform

  const float* Arow = A + (size_t)(m0 + lm) * lda;
  const _Float16* Bbase = Wh + (size_t)(n0 + lm) * Kpad;  // tile j at +j*16*Kpad

  v8f acc[4] = {v8f{}, v8f{}, v8f{}, v8f{}};

  if (ntiles == 4) {                         // fast path: all heavy GEMMs
    int k0 = 0;
    for (; k0 + 64 <= Kpad; k0 += 64) {
      __builtin_prefetch(Arow + k0 + 64, 0, 1);
#pragma unroll
      for (int kk = 0; kk < 2; ++kk) {
        const v16h av = load_a(Arow, k0 + kk * 32, hi);
#pragma unroll
        for (int j = 0; j < 4; ++j)
          acc[j] = wmma16(av, load_b(Bbase + (size_t)j * 16 * Kpad, k0 + kk * 32, hi), acc[j]);
      }
    }
    if (k0 < Kpad) {
      const v16h av = load_a(Arow, k0, hi);
#pragma unroll
      for (int j = 0; j < 4; ++j)
        acc[j] = wmma16(av, load_b(Bbase + (size_t)j * 16 * Kpad, k0, hi), acc[j]);
    }
  } else {                                   // ragged Npad (48, 16)
    for (int k0 = 0; k0 < Kpad; k0 += 32) {
      const v16h av = load_a(Arow, k0, hi);
#pragma unroll
      for (int j = 0; j < 4; ++j)
        if (j < ntiles)
          acc[j] = wmma16(av, load_b(Bbase + (size_t)j * 16 * Kpad, k0, hi), acc[j]);
    }
  }

  // epilogue (stores guarded for ragged N)
#pragma unroll
  for (int j = 0; j < 4; ++j) {
    if (j >= ntiles) break;
    const int bcol = n0 + j * 16 + lm;
    if (bcol < N) {
      const float badd = bias ? bias[bcol] : 0.0f;
      const float gmul = bng ? bng[bcol] * bninv : 1.0f;
      const float gsh  = bnb ? bnb[bcol] : 0.0f;
#pragma unroll
      for (int r = 0; r < 8; ++r) {
        const int row = m0 + r + hi * 8;     // C/D: VGPR r -> M = r (+8 for hi)
        float v = acc[j][r] + badd;
        if (bng) v = v * gmul + gsh;
        if (act == 1)      v = fmaxf(v, 0.0f);
        else if (act == 2) v = (v > 20.0f) ? v : log1pf(__expf(v));
        else if (act == 3) v = sigf(v);
        C[(size_t)row * ldc + bcol] = v;
      }
    }
  }
}

// ---------------------------------------------------------------------------
// Feature extraction: spectral conv3d (spk,1,1) + BN + relu (into LDS), then
// depthwise spatial conv (1,sk,sk) + BN + relu, max over depth. One block per
// (batch, out-channel).
// ---------------------------------------------------------------------------
__global__ void fe3d_kernel(const float* __restrict__ x,
                            const float* __restrict__ w1, const float* __restrict__ cb1,
                            const float* __restrict__ g1, const float* __restrict__ b1,
                            const float* __restrict__ w2, const float* __restrict__ cb2,
                            const float* __restrict__ g2, const float* __restrict__ b2,
                            float* __restrict__ fc, int spk, int sk, int sbase) {
  __shared__ float sh[CIN * HW];
  const int b = blockIdx.x, oc = blockIdx.y, tid = threadIdx.x;
  const float inv = 0.9999950000374997f;  // 1/sqrt(1+1e-5)
  const int pad = spk >> 1;
  const float sc1 = g1[oc] * inv, of1 = b1[oc], cbv1 = cb1[oc];

  for (int idx = tid; idx < CIN * HW; idx += blockDim.x) {
    const int d = idx / HW, hw = idx % HW;
    float acc = cbv1;
    for (int kd = 0; kd < spk; ++kd) {
      const int di = d + kd - pad;
      if (di >= 0 && di < CIN)
        acc += x[((size_t)(b * CIN + di)) * HW + hw] * w1[oc * spk + kd];
    }
    sh[idx] = fmaxf(acc * sc1 + of1, 0.0f);
  }
  __syncthreads();

  const int pad2 = sk >> 1;
  const float sc2 = g2[oc] * inv, of2 = b2[oc], cbv2 = cb2[oc];
  for (int hw = tid; hw < HW; hw += blockDim.x) {
    const int h = hw / HH, w = hw % HH;
    float best = -1e30f;
    for (int d = 0; d < CIN; ++d) {
      float acc = cbv2;
      for (int i = 0; i < sk; ++i) {
        const int hh = h + i - pad2;
        if (hh < 0 || hh >= HH) continue;
        for (int j = 0; j < sk; ++j) {
          const int ww = w + j - pad2;
          if (ww < 0 || ww >= HH) continue;
          acc += sh[d * HW + hh * HH + ww] * w2[(oc * sk + i) * sk + j];
        }
      }
      best = fmaxf(best, fmaxf(acc * sc2 + of2, 0.0f));
    }
    fc[((size_t)b * HW + hw) * EE + sbase + oc] = best;
  }
}

// depthwise 3x3 conv2d (into LDS) -> pointwise 16x30 + BN + relu. 1 block/batch.
__global__ void fe2d_kernel(const float* __restrict__ x,
                            const float* __restrict__ dww, const float* __restrict__ dwb,
                            const float* __restrict__ pww, const float* __restrict__ pwb,
                            const float* __restrict__ g3, const float* __restrict__ b3,
                            float* __restrict__ fc, int sbase) {
  __shared__ float sh[CIN * HW];
  const int b = blockIdx.x, tid = threadIdx.x;
  const float inv = 0.9999950000374997f;

  for (int idx = tid; idx < CIN * HW; idx += blockDim.x) {
    const int c = idx / HW, hw = idx % HW, h = hw / HH, w = hw % HH;
    float acc = dwb[c];
    for (int i = 0; i < 3; ++i) {
      const int hh = h + i - 1;
      if (hh < 0 || hh >= HH) continue;
      for (int j = 0; j < 3; ++j) {
        const int ww = w + j - 1;
        if (ww < 0 || ww >= HH) continue;
        acc += x[((size_t)(b * CIN + c)) * HW + hh * HH + ww] * dww[c * 9 + i * 3 + j];
      }
    }
    sh[idx] = acc;
  }
  __syncthreads();

  for (int idx = tid; idx < 16 * HW; idx += blockDim.x) {
    const int oc = idx / HW, hw = idx % HW;
    float acc = pwb[oc];
    for (int c = 0; c < CIN; ++c) acc += sh[c * HW + hw] * pww[oc * CIN + c];
    acc = fmaxf(acc * g3[oc] * inv + b3[oc], 0.0f);
    fc[((size_t)b * HW + hw) * EE + sbase + 16 + oc] = acc;
  }
}

// LayerNorm over last dim E (one thread per row — rows are L2-hot).
__global__ void ln_kernel(const float* __restrict__ in, float* __restrict__ out,
                          const float* __restrict__ g, const float* __restrict__ bb,
                          float eps, int M, int E) {
  const int row = blockIdx.x * blockDim.x + threadIdx.x;
  if (row >= M) return;
  const float* p = in + (size_t)row * E;
  float mu = 0.0f;
  for (int j = 0; j < E; ++j) mu += p[j];
  mu /= E;
  float var = 0.0f;
  for (int j = 0; j < E; ++j) { const float d = p[j] - mu; var += d * d; }
  const float r = rsqrtf(var / E + eps);
  float* o = out + (size_t)row * E;
  for (int j = 0; j < E; ++j) o[j] = (p[j] - mu) * r * g[j] + bb[j];
}

// x2[:,i] = x[:, i<L-1 ? L-2-i : L-1]
__global__ void reverse_kernel(const float* __restrict__ in, float* __restrict__ out, int n) {
  const int idx = blockIdx.x * blockDim.x + threadIdx.x;
  if (idx >= n) return;
  const int e = idx % EE, bl = idx / EE, l = bl % LL, b = bl / LL;
  const int rl = (l < LL - 1) ? (LL - 2 - l) : (LL - 1);
  out[idx] = in[((size_t)b * LL + rl) * EE + e];
}

// causal depthwise conv1d (k=4, left pad 3) + bias + SiLU; input = xz[..., :256]
__global__ void conv1d_silu_kernel(const float* __restrict__ xz,
                                   const float* __restrict__ cw,
                                   const float* __restrict__ cb,
                                   float* __restrict__ u, int n) {
  const int idx = blockIdx.x * blockDim.x + threadIdx.x;
  if (idx >= n) return;
  const int d = idx % DI, bl = idx / DI, l = bl % LL, b = bl / LL;
  float acc = cb[d];
#pragma unroll
  for (int j = 0; j < 4; ++j) {
    const int li = l - 3 + j;
    if (li >= 0) acc += xz[((size_t)b * LL + li) * (2 * DI) + d] * cw[d * 4 + j];
  }
  u[idx] = acc * sigf(acc);
}

// selective scan: one block per batch sample, thread = channel d, h in regs.
__global__ void scan_kernel(const float* __restrict__ dt, const float* __restrict__ u,
                            const float* __restrict__ dbl, const float* __restrict__ Alog,
                            const float* __restrict__ Dp, float* __restrict__ y) {
  const int b = blockIdx.x, d = threadIdx.x;
  float Av[DSTATE], h[DSTATE];
#pragma unroll
  for (int nn = 0; nn < DSTATE; ++nn) { Av[nn] = -__expf(Alog[d * DSTATE + nn]); h[nn] = 0.0f; }
  const float Dd = Dp[d];
  for (int l = 0; l < LL; ++l) {
    const size_t t = (size_t)b * LL + l;
    const float dtv = dt[t * DI + d];
    const float uv  = u[t * DI + d];
    const float du  = dtv * uv;
    const float* bc = dbl + t * 40;
    float ys = 0.0f;
#pragma unroll
    for (int nn = 0; nn < DSTATE; ++nn) {
      const float dA = __expf(dtv * Av[nn]);
      h[nn] = dA * h[nn] + du * bc[8 + nn];
      ys += h[nn] * bc[24 + nn];
    }
    y[t * DI + d] = ys + uv * Dd;
  }
}

// yz = y * silu(z), z = xz[..., 256:]
__global__ void ygate_kernel(const float* __restrict__ y, const float* __restrict__ xz,
                             float* __restrict__ yz, int n) {
  const int idx = blockIdx.x * blockDim.x + threadIdx.x;
  if (idx >= n) return;
  const int d = idx % DI;
  const size_t t = idx / DI;
  const float zv = xz[t * (2 * DI) + DI + d];
  yz[idx] = y[idx] * zv * sigf(zv);
}

// xio = 0.5*(x1 + reverse(x3f)) + xio
__global__ void combine_kernel(float* __restrict__ xio, const float* __restrict__ x1,
                               const float* __restrict__ x3f, int n) {
  const int idx = blockIdx.x * blockDim.x + threadIdx.x;
  if (idx >= n) return;
  const int e = idx % EE, bl = idx / EE, l = bl % LL, b = bl / LL;
  const int rl = (l < LL - 1) ? (LL - 2 - l) : (LL - 1);
  xio[idx] = 0.5f * (x1[idx] + x3f[((size_t)b * LL + rl) * EE + e]) + xio[idx];
}

// out[b*ostride + ooff + e] = mean_l in[b,l,e]
__global__ void mean_l_kernel(const float* __restrict__ in, float* __restrict__ out,
                              int BE, int E, int ostride, int ooff) {
  const int idx = blockIdx.x * blockDim.x + threadIdx.x;
  if (idx >= BE) return;
  const int b = idx / E, e = idx % E;
  float s = 0.0f;
  for (int l = 0; l < LL; ++l) s += in[((size_t)b * LL + l) * E + e];
  out[(size_t)b * ostride + ooff + e] = s * (1.0f / LL);
}

__global__ void halfsum_kernel(const float* a, const float* b, float* o, int n) {
  const int i = blockIdx.x * blockDim.x + threadIdx.x;
  if (i < n) o[i] = 0.5f * (a[i] + b[i]);
}

__global__ void copy_kernel(const float* a, float* o, int n) {
  const int i = blockIdx.x * blockDim.x + threadIdx.x;
  if (i < n) o[i] = a[i];
}

// x = x*sig[b,e] + id
__global__ void sigres_kernel(float* __restrict__ x, const float* __restrict__ sg,
                              const float* __restrict__ idb, int n) {
  const int idx = blockIdx.x * blockDim.x + threadIdx.x;
  if (idx >= n) return;
  const int e = idx % EE, b = idx / (LL * EE);
  x[idx] = x[idx] * sg[b * EE + e] + idb[idx];
}

// fused = softmax(logits)[0]*spa + [1]*spe
__global__ void fusemix_kernel(const float* __restrict__ spa, const float* __restrict__ spe,
                               const float* __restrict__ lg, float* __restrict__ out, int n) {
  const int idx = blockIdx.x * blockDim.x + threadIdx.x;
  if (idx >= n) return;
  const int b = idx / (LL * EE);
  const float l0 = lg[b * 2], l1 = lg[b * 2 + 1];
  const float m = fmaxf(l0, l1);
  const float e0 = __expf(l0 - m), e1 = __expf(l1 - m);
  const float s = e0 + e1;
  out[idx] = (e0 / s) * spa[idx] + (e1 / s) * spe[idx];
}

// ---------------------------------------------------------------------------
static inline int cdiv(int a, int b) { return (a + b - 1) / b; }

struct PW { const _Float16* p; int Kpad; int Npad; };

extern "C" void kernel_launch(void* const* d_in, const int* in_sizes, int n_in,
                              void* d_out, int out_size, void* d_ws, size_t ws_size,
                              hipStream_t stream) {
  (void)in_sizes; (void)n_in; (void)out_size; (void)ws_size;
  auto F = [&](int i) { return (const float*)d_in[i]; };
  const float BNINV = 0.9999950000374997f;

  // --- workspace carve-up (256B-aligned chunks) ---
  char* base = (char*)d_ws;
  size_t off = 0;
  auto allocB = [&](size_t bytes) {
    void* p = base + off;
    off = (off + bytes + 255) & ~(size_t)255;
    return p;
  };
  auto allocF = [&](size_t n) { return (float*)allocB(n * sizeof(float)); };

  float* fc    = allocF((size_t)BL * EE);   // reused as "fused" at the end
  float* spa   = allocF((size_t)BL * EE);
  float* spe   = allocF((size_t)BL * EE);
  float* idspa = allocF((size_t)BL * EE);
  float* idspe = allocF((size_t)BL * EE);
  float* xl    = allocF((size_t)BL * EE);
  float* x2    = allocF((size_t)BL * EE);
  float* x1o   = allocF((size_t)BL * EE);
  float* x3f   = allocF((size_t)BL * EE);
  float* xz    = allocF((size_t)BL * 2 * DI);
  float* ubuf  = allocF((size_t)BL * DI);
  float* dblb  = allocF((size_t)BL * 40);
  float* dtb   = allocF((size_t)BL * DI);
  float* ybuf  = allocF((size_t)BL * DI);
  float* yzb   = allocF((size_t)BL * DI);
  float* msa   = allocF((size_t)BB * EE);
  float* msb   = allocF((size_t)BB * EE);
  float* pooled= allocF((size_t)BB * EE);
  float* sigb  = allocF((size_t)BB * EE);
  float* gfb   = allocF((size_t)BB * 2 * EE);
  float* hidb  = allocF((size_t)BB * 64);
  float* lgb   = allocF((size_t)BB * 2);
  float* fmean = allocF((size_t)BB * EE);

  // --- pre-pack all GEMM weights to padded f16 (runs on stream, ~3MB total) --
  auto packW = [&](const float* Wsrc, int N, int K) -> PW {
    const int Npad = (N + 15) & ~15, Kpad = (K + 31) & ~31;
    _Float16* dst = (_Float16*)allocB((size_t)Npad * Kpad * sizeof(_Float16));
    const int total = Npad * Kpad;
    pack_w_f16<<<dim3(cdiv(total, 256)), 256, 0, stream>>>(Wsrc, dst, N, K, Kpad, total);
    return {dst, Kpad, Npad};
  };

  PW pw_fu   = packW(F(144), EE, EE);
  PW pw_af1  = packW(F(2), 64, 2 * EE);
  PW pw_af2  = packW(F(3), 2, 64);
  PW pw_head = packW(F(146), NC, EE);
  PW pw_sc[3], pw_fuse[3], pw_in[6], pw_xp[6], pw_dt[6], pw_op[6];
  for (int blk = 0; blk < 3; ++blk) {
    const int bb = 4 + 27 * blk;
    pw_sc[blk]   = packW(F(bb + 4), EE, EE);
    pw_fuse[blk] = packW(F(bb + 0), EE, EE);
    for (int br = 0; br < 2; ++br) {
      const int sb = bb + (br == 0 ? 5 : 16);
      const int i = blk * 2 + br;
      pw_in[i] = packW(F(sb + 6), 2 * DI, EE);     // (512,128)
      pw_xp[i] = packW(F(sb + 10), 40, DI);        // (40,256) -> Npad 48
      pw_dt[i] = packW(F(sb + 5), DI, DTRANK);     // (256,8)  -> Kpad 32
      pw_op[i] = packW(F(sb + 9), EE, DI);         // (128,256)
    }
  }

  auto gemm = [&](const float* Aq, int lda, const PW& wq, const float* bias,
                  const float* g, const float* bbp, float* Cq, int ldc,
                  int M, int N, int act) {
    dim3 grid(cdiv(wq.Npad, 64), cdiv(M, 64)), block(32, 4);
    gemm_wmma_f16<<<grid, block, 0, stream>>>(Aq, lda, wq.p, wq.Kpad, bias, g, bbp,
                                              BNINV, Cq, ldc, M, N, wq.Npad, act);
  };
  auto ew = [&](int n) { return dim3(cdiv(n, 256)); };

  // ---- feature extraction (4 scales) ----
  const int SKS[4] = {3, 5, 7, 9}, SPKS[4] = {5, 7, 9, 11};
  const float* xin = F(149);
  for (int s = 0; s < 4; ++s) {
    const int fb = 85 + 14 * s;
    fe3d_kernel<<<dim3(BB, 16), 256, 0, stream>>>(
        xin, F(fb + 7), F(fb + 6), F(fb + 1), F(fb + 0),
        F(fb + 9), F(fb + 8), F(fb + 3), F(fb + 2),
        fc, SPKS[s], SKS[s], s * 32);
    fe2d_kernel<<<dim3(BB), 256, 0, stream>>>(
        xin, F(fb + 11), F(fb + 10), F(fb + 13), F(fb + 12),
        F(fb + 5), F(fb + 4), fc, s * 32);
  }

  // ---- fusion conv (1x1) + BN + relu -> tokens ----
  gemm(fc, EE, pw_fu, F(141), F(143), F(142), spa, EE, BL, EE, /*relu*/1);
  copy_kernel<<<ew(BL * EE), 256, 0, stream>>>(spa, spe, BL * EE);

  // ---- ssm pipeline ----
  auto run_ssm = [&](const float* xi, float* xo, int sb, int wi) {
    gemm(xi, EE, pw_in[wi], nullptr, nullptr, nullptr, xz, 2 * DI, BL, 2 * DI, 0);
    conv1d_silu_kernel<<<ew(BL * DI), 256, 0, stream>>>(xz, F(sb + 3), F(sb + 2), ubuf, BL * DI);
    gemm(ubuf, DI, pw_xp[wi], nullptr, nullptr, nullptr, dblb, 40, BL, 40, 0);
    gemm(dblb, 40, pw_dt[wi], F(sb + 4), nullptr, nullptr, dtb, DI, BL, DI, /*softplus*/2);
    scan_kernel<<<dim3(BB), DI, 0, stream>>>(dtb, ubuf, dblb, F(sb + 0), F(sb + 1), ybuf);
    ygate_kernel<<<ew(BL * DI), 256, 0, stream>>>(ybuf, xz, yzb, BL * DI);
    gemm(yzb, DI, pw_op[wi], nullptr, nullptr, nullptr, xo, EE, BL, EE, 0);
  };

  // ---- residual blocks ----
  for (int blk = 0; blk < 3; ++blk) {
    const int bb = 4 + 27 * blk;
    gemm(spa, EE, pw_sc[blk], F(bb + 1), nullptr, nullptr, idspa, EE, BL, EE, 0);
    ln_kernel<<<ew(BL), 256, 0, stream>>>(idspa, idspa, F(bb + 3), F(bb + 2), 1e-5f, BL, EE);
    gemm(spe, EE, pw_sc[blk], F(bb + 1), nullptr, nullptr, idspe, EE, BL, EE, 0);
    ln_kernel<<<ew(BL), 256, 0, stream>>>(idspe, idspe, F(bb + 3), F(bb + 2), 1e-5f, BL, EE);

    for (int br = 0; br < 2; ++br) {           // 0: spa, 1: spe
      float* xb = (br == 0) ? spa : spe;
      const int sb = bb + (br == 0 ? 5 : 16);
      const int wi = blk * 2 + br;
      ln_kernel<<<ew(BL), 256, 0, stream>>>(xb, xl, F(sb + 8), F(sb + 7), 1e-6f, BL, EE);
      run_ssm(xl, x1o, sb, wi);                // forward direction
      reverse_kernel<<<ew(BL * EE), 256, 0, stream>>>(xl, x2, BL * EE);
      run_ssm(x2, x3f, sb, wi);                // reverse direction
      combine_kernel<<<ew(BL * EE), 256, 0, stream>>>(xb, x1o, x3f, BL * EE);
    }

    // cross-branch sigmoid fusion
    mean_l_kernel<<<ew(BB * EE), 256, 0, stream>>>(spa, msa, BB * EE, EE, EE, 0);
    mean_l_kernel<<<ew(BB * EE), 256, 0, stream>>>(spe, msb, BB * EE, EE, EE, 0);
    halfsum_kernel<<<ew(BB * EE), 256, 0, stream>>>(msa, msb, pooled, BB * EE);
    gemm(pooled, EE, pw_fuse[blk], nullptr, nullptr, nullptr, sigb, EE, BB, EE, /*sigmoid*/3);
    sigres_kernel<<<ew(BL * EE), 256, 0, stream>>>(spa, sigb, idspa, BL * EE);
    sigres_kernel<<<ew(BL * EE), 256, 0, stream>>>(spe, sigb, idspe, BL * EE);
  }

  // ---- adaptive fusion head ----
  mean_l_kernel<<<ew(BB * EE), 256, 0, stream>>>(spa, gfb, BB * EE, EE, 2 * EE, 0);
  mean_l_kernel<<<ew(BB * EE), 256, 0, stream>>>(spe, gfb, BB * EE, EE, 2 * EE, EE);
  gemm(gfb, 2 * EE, pw_af1, F(0), nullptr, nullptr, hidb, 64, BB, 64, /*relu*/1);
  gemm(hidb, 64, pw_af2, F(1), nullptr, nullptr, lgb, 2, BB, 2, 0);
  fusemix_kernel<<<ew(BL * EE), 256, 0, stream>>>(spa, spe, lgb, fc, BL * EE);
  ln_kernel<<<ew(BL), 256, 0, stream>>>(fc, fc, F(148), F(147), 1e-5f, BL, EE);
  mean_l_kernel<<<ew(BB * EE), 256, 0, stream>>>(fc, fmean, BB * EE, EE, EE, 0);
  gemm(fmean, EE, pw_head, F(145), nullptr, nullptr, (float*)d_out, NC, BB, NC, 0);
}
